// AttentionBlock_74775380623444
// MI455X (gfx1250) — matrix operand order repaired and checked by reading, not compiled, not run on hardware
//
#include <hip/hip_runtime.h>

// ---------------------------------------------------------------------------
// MI455X (gfx1250) wave32 WMMA implementation of the windowed-attention block.
// All GEMMs (QKV, Q@K^T, P@V, out-proj) use v_wmma_f32_16x16x32_f16.
// The QKV/out-proj GEMM is double-buffered; the A tile is fetched by the
// Tensor Data Mover (tensor_load_to_lds + s_wait_tensorcnt) when available.
// ---------------------------------------------------------------------------

typedef __attribute__((ext_vector_type(16))) _Float16 v16h;
typedef __attribute__((ext_vector_type(8)))  float    v8f;

__device__ __forceinline__ int laneid() { return threadIdx.x & 31; }

__device__ __forceinline__ v8f wmma_f16(v16h a, v16h b, v8f c) {
  // 8 args: (neg_a, A, neg_b, B, c_mod, C, reuse_a, reuse_b)
  return __builtin_amdgcn_wmma_f32_16x16x32_f16(false, a, false, b, (short)0, c,
                                                false, false);
}

// A fragment: 16x32 f16, row-major source, leading dim ld (in halfs).
__device__ __forceinline__ v16h frag_a(const _Float16* p, int ld) {
  int lane = laneid(); int r = lane & 15; int hh = lane >> 4;
  const _Float16* row = p + r * ld;
  v16h a;
#pragma unroll
  for (int j = 0; j < 8; ++j) {
    int k = ((j & 4) << 2) + hh * 8 + ((j & 3) << 1);
    a[2 * j]     = row[k];
    a[2 * j + 1] = row[k + 1];
  }
  return a;
}

// B fragment: 32x16 f16, source stored "N-major": p[c*ld + k] = B[k][c].
__device__ __forceinline__ v16h frag_b(const _Float16* p, int ld) {
  int lane = laneid(); int c = lane & 15; int hh = lane >> 4;
  const _Float16* col = p + c * ld;
  v16h b;
#pragma unroll
  for (int j = 0; j < 8; ++j) {
    int k = ((j & 4) << 2) + hh * 8 + ((j & 3) << 1);
    b[2 * j]     = col[k];
    b[2 * j + 1] = col[k + 1];
  }
  return b;
}

// ---------------------------------------------------------------------------
// Tensor Data Mover path (guarded; manual-load fallback otherwise).
// ---------------------------------------------------------------------------
#if defined(__has_builtin)
#if __has_builtin(__builtin_amdgcn_tensor_load_to_lds) && \
    __has_builtin(__builtin_amdgcn_s_wait_tensorcnt)
#define USE_TDM 1
#endif
#endif

#ifdef USE_TDM
typedef unsigned int u32x4 __attribute__((ext_vector_type(4)));
typedef int          i32x4 __attribute__((ext_vector_type(4)));
typedef int          i32x8 __attribute__((ext_vector_type(8)));

// DMA one 128x32-half tile of an 8192x1024 f16 tensor into LDS.
// pad_interval = 16 DWORDs (one 32-half row), pad_amount = 4 DWORDs
// => TDM writes the tile with leading dim 40 halfs, matching frag_a/frag_b.
__device__ __forceinline__ void tdm_load_tile_a(const _Float16* gsrc,
                                                unsigned lds_byte_off) {
  unsigned long long ga = (unsigned long long)(uintptr_t)gsrc;
  u32x4 g0;
  g0.x = 1u;                                            // count=1, user D#
  g0.y = lds_byte_off;                                  // lds_addr
  g0.z = (unsigned)ga;                                  // global_addr[31:0]
  g0.w = (unsigned)((ga >> 32) & 0x01ffffffu) | (2u << 30); // addr[56:32]|type=2
  i32x8 g1;
  g1[0] = (1 << 16) | (1 << 20) | (3 << 22) | (3 << 25); // 2B elems, pad 16DW/4DW
  g1[1] = (int)(1024u << 16);                            // tensor_dim0 = 1024
  g1[2] = (int)(8192u << 16);                            // tensor_dim1 = 8192
  g1[3] = (int)(32u << 16);                              // tile_dim0 = 32
  g1[4] = 128;                                           // tile_dim1 = 128
  g1[5] = 1024;                                          // dim0_stride = 1024
  g1[6] = 0;
  g1[7] = 0;
  i32x4 z4 = {0, 0, 0, 0};
#if __clang_major__ >= 23
  i32x8 z8 = {0, 0, 0, 0, 0, 0, 0, 0};
  __builtin_amdgcn_tensor_load_to_lds(g0, g1, z4, z4, z8, 0);
#else
  __builtin_amdgcn_tensor_load_to_lds(g0, g1, z4, z4, 0);
#endif
}
#endif

// ---------------------------------------------------------------------------
// 1) LayerNorm (fp32 in, f16 out)
// ---------------------------------------------------------------------------
__global__ __launch_bounds__(256) void ln_kernel(const float* __restrict__ x,
                                                 const float* __restrict__ gamma,
                                                 _Float16* __restrict__ xn) {
  __shared__ float rs[256], rss[256];
  int row = blockIdx.x;
  int tid = threadIdx.x;
  const float* xr = x + (long)row * 1024;
  float s = 0.f, ss = 0.f;
#pragma unroll
  for (int u = 0; u < 4; ++u) {
    float v = xr[tid + u * 256];
    s += v; ss += v * v;
  }
  rs[tid] = s; rss[tid] = ss;
  __syncthreads();
  for (int str = 128; str > 0; str >>= 1) {
    if (tid < str) { rs[tid] += rs[tid + str]; rss[tid] += rss[tid + str]; }
    __syncthreads();
  }
  float mu   = rs[0] * (1.0f / 1024.0f);
  float var  = rss[0] * (1.0f / 1024.0f) - mu * mu;
  float rstd = rsqrtf(var + 1e-5f);
  _Float16* xo = xn + (long)row * 1024;
#pragma unroll
  for (int u = 0; u < 4; ++u) {
    int idx = tid + u * 256;
    xo[idx] = (_Float16)((xr[idx] - mu) * rstd * gamma[idx]);
  }
}

// ---------------------------------------------------------------------------
// 2) fp32 -> f16 weight conversion
// ---------------------------------------------------------------------------
__global__ void cvt_kernel(const float* __restrict__ s, _Float16* __restrict__ d, int n) {
  for (int i = blockIdx.x * blockDim.x + threadIdx.x; i < n; i += gridDim.x * blockDim.x)
    d[i] = (_Float16)s[i];
}

// ---------------------------------------------------------------------------
// 3/5) Double-buffered WMMA GEMM: C(8192 x N) = A(8192x1024,f16) * Bw(1024xN,f16)
//      BM=BN=128, BK=32, 256 threads (8 waves, 4x2 wave grid, 32x64 per wave).
//      A tile via TDM (wave 0 issues, waits TENSORcnt post-compute);
//      B tile via coalesced reads + register->LDS transpose.
//      mode 0: QKV epilogue (scatter q/k/v f16 tiles + fp32 memories)
//      mode 1: plain fp32 store (output projection)
// ---------------------------------------------------------------------------
__global__ __launch_bounds__(256) void gemm_kernel(
    const _Float16* __restrict__ A, const _Float16* __restrict__ Bw,
    int N, int mode,
    _Float16* __restrict__ qb, _Float16* __restrict__ kb, _Float16* __restrict__ vb,
    float* __restrict__ mem, float* __restrict__ outf) {
  __shared__ _Float16 Alds[2][128 * 40];   // [m][k], ld=40
  __shared__ _Float16 Blds[2][128 * 40];   // transposed [n][k], ld=40
  int tid = threadIdx.x;
  int N0 = blockIdx.x * 128;
  int M0 = blockIdx.y * 128;
  int wv = tid >> 5;
  int wm = wv >> 1, wn = wv & 1;
  int bkk = tid >> 3, bseg = tid & 7;      // B-tile loader indices
#ifndef USE_TDM
  int arow = tid >> 1, aseg = tid & 1;     // A-tile loader indices (fallback)
#endif

  v8f acc[2][4];
#pragma unroll
  for (int i = 0; i < 2; ++i)
#pragma unroll
    for (int j = 0; j < 4; ++j)
#pragma unroll
      for (int e = 0; e < 8; ++e) acc[i][j][e] = 0.f;

  // ---- prologue: stage k-step 0 into buffer 0 ----
  {
    const _Float16* bsrc = Bw + (long)bkk * N + N0 + bseg * 16;
    uint4 b0 = *(const uint4*)bsrc;
    uint4 b1 = *(const uint4*)(bsrc + 8);
#ifdef USE_TDM
    if (tid < 32)
      tdm_load_tile_a(A + (long)M0 * 1024, (unsigned)(uintptr_t)&Alds[0][0]);
#else
    const uint4* as = (const uint4*)(A + (long)(M0 + arow) * 1024 + aseg * 16);
    uint4 a0 = as[0], a1 = as[1];
    uint4* ad = (uint4*)(&Alds[0][arow * 40 + aseg * 16]);
    ad[0] = a0; ad[1] = a1;
#endif
    _Float16 tb[16];
    *(uint4*)&tb[0] = b0; *(uint4*)&tb[8] = b1;
#pragma unroll
    for (int u = 0; u < 16; ++u)
      Blds[0][(bseg * 16 + u) * 40 + bkk] = tb[u];
#ifdef USE_TDM
    if (tid < 32) __builtin_amdgcn_s_wait_tensorcnt(0);
#endif
  }
  __syncthreads();

  // ---- pipelined main loop: prefetch it+1 while computing it ----
  for (int it = 0; it < 32; ++it) {
    int p = it & 1;
    bool more = (it < 31);
    int k0n = (it + 1) * 32;
    uint4 b0, b1;
#ifndef USE_TDM
    uint4 a0, a1;
#endif
    if (more) {
      const _Float16* bsrc = Bw + (long)(k0n + bkk) * N + N0 + bseg * 16;
      b0 = *(const uint4*)bsrc;
      b1 = *(const uint4*)(bsrc + 8);
#ifdef USE_TDM
      if (tid < 32)
        tdm_load_tile_a(A + (long)M0 * 1024 + k0n,
                        (unsigned)(uintptr_t)&Alds[p ^ 1][0]);
#else
      const uint4* as =
          (const uint4*)(A + (long)(M0 + arow) * 1024 + k0n + aseg * 16);
      a0 = as[0]; a1 = as[1];
#endif
    }

    // compute on buffer p
    v16h afr[2];
#pragma unroll
    for (int mt = 0; mt < 2; ++mt)
      afr[mt] = frag_a(&Alds[p][(wm * 32 + mt * 16) * 40], 40);
#pragma unroll
    for (int nt = 0; nt < 4; ++nt) {
      v16h bfr = frag_b(&Blds[p][(wn * 64 + nt * 16) * 40], 40);
#pragma unroll
      for (int mt = 0; mt < 2; ++mt)
        acc[mt][nt] = wmma_f16(afr[mt], bfr, acc[mt][nt]);
    }

    if (more) {
      _Float16 tb[16];
      *(uint4*)&tb[0] = b0; *(uint4*)&tb[8] = b1;
#pragma unroll
      for (int u = 0; u < 16; ++u)
        Blds[p ^ 1][(bseg * 16 + u) * 40 + bkk] = tb[u];
#ifdef USE_TDM
      if (tid < 32) __builtin_amdgcn_s_wait_tensorcnt(0);
#else
      uint4* ad = (uint4*)(&Alds[p ^ 1][arow * 40 + aseg * 16]);
      ad[0] = a0; ad[1] = a1;
#endif
    }
    __syncthreads();
  }

  // ---- epilogue ----
  int lane = laneid(); int cN = lane & 15; int hh = lane >> 4;
#pragma unroll
  for (int mt = 0; mt < 2; ++mt)
#pragma unroll
    for (int nt = 0; nt < 4; ++nt)
#pragma unroll
      for (int e = 0; e < 8; ++e) {
        int row = M0 + wm * 32 + mt * 16 + e + 8 * hh;
        int col = N0 + wn * 64 + nt * 16 + cN;
        float val = acc[mt][nt][e];
        if (mode == 0) {
          int which = col >> 10;           // 0=q 1=k 2=v
          int hd = col & 1023;
          int h = hd >> 6, d = hd & 63;
          int bb = row >> 12;
          int pos = row & 4095;
          int ww = pos >> 9, ii = pos & 511;
          long off = ((((long)(bb * 8 + ww)) * 16 + h) << 15) + ii * 64 + d;
          _Float16 hv = (_Float16)val;
          if (which == 0)      qb[off] = hv;
          else if (which == 1) kb[off] = hv;
          else                 vb[off] = hv;
          if (which >= 1 && ww == 7) {     // memories = (k,v) of last block, fp32
            long moff = ((((long)((which - 1) * 2 + bb)) * 16 + h) << 15) + ii * 64 + d;
            mem[moff] = val;
          }
        } else {
          outf[(long)row * N + col] = val;
        }
      }
}

// ---------------------------------------------------------------------------
// 4) Flash-style windowed attention. One block per (b, w, h, half); 512
//    threads = 16 waves; each wave owns one 16-row q tile. K window =
//    [prev block (or xl_memories, fp32), current block], 1024 cols, 32/step.
// ---------------------------------------------------------------------------
__global__ __launch_bounds__(512) void attn_kernel(
    const _Float16* __restrict__ qb, const _Float16* __restrict__ kb,
    const _Float16* __restrict__ vb, const float* __restrict__ xlmem,
    const float* __restrict__ bias, _Float16* __restrict__ aout) {
  __shared__ _Float16 Kt[32 * 72];        // [j][d], ld=72
  __shared__ _Float16 Vt[64 * 40];        // transposed [d][j], ld=40
  __shared__ _Float16 Pl[16 * 16 * 40];   // per-wave P tiles, ld=40

  int blk = blockIdx.x;
  int half = blk & 1;
  int h = (blk >> 1) & 15;
  int w = (blk >> 5) & 7;
  int b = blk >> 8;
  int tid = threadIdx.x;
  int wv = tid >> 5;
  int lane = tid & 31;
  int cN = lane & 15, hh = lane >> 4;

  long bwh = (((long)(b * 8 + w)) * 16 + h) << 15;
  const _Float16* Q  = qb + bwh;
  const _Float16* Kc = kb + bwh;
  const _Float16* Vc = vb + bwh;
  long pwh = (w > 0) ? ((((long)(b * 8 + w - 1)) * 16 + h) << 15) : 0;
  const _Float16* Kp16 = kb + pwh;
  const _Float16* Vp16 = vb + pwh;
  const float* Kp32 = xlmem + (((long)(b * 16 + h)) << 15);
  const float* Vp32 = xlmem + (((long)((2 + b) * 16 + h)) << 15);

  int i0 = half * 256 + wv * 16;
  v16h qa0 = frag_a(Q + (long)i0 * 64, 64);
  v16h qa1 = frag_a(Q + (long)i0 * 64 + 32, 64);

  float rm[8], rl[8];
  v8f o[4];
#pragma unroll
  for (int e = 0; e < 8; ++e) { rm[e] = -3.0e38f; rl[e] = 0.f; }
#pragma unroll
  for (int dt = 0; dt < 4; ++dt)
#pragma unroll
    for (int e = 0; e < 8; ++e) o[dt][e] = 0.f;

  int nsteps = half ? 32 : 24;       // causal window: half 0 never sees j>=768
  const float* bh = bias + (long)h * 512 * 1024;
  const float scale = 0.125f;        // 64^-0.5

  for (int s = 0; s < nsteps; ++s) {
    int j0 = s * 32;
    __syncthreads();
    // cooperative load of K tile (row-major) and V tile (transposed)
#pragma unroll
    for (int u = 0; u < 4; ++u) {
      int idx = tid + u * 512;        // 0..2047 over 32x64 tile
      int jj = idx >> 6, dd = idx & 63;
      int jg = j0 + jj;
      _Float16 kvv, vvv;
      if (jg < 512) {
        if (w > 0) { kvv = Kp16[jg * 64 + dd]; vvv = Vp16[jg * 64 + dd]; }
        else       { kvv = (_Float16)Kp32[jg * 64 + dd]; vvv = (_Float16)Vp32[jg * 64 + dd]; }
      } else {
        int jr = jg - 512;
        kvv = Kc[jr * 64 + dd];
        vvv = Vc[jr * 64 + dd];
      }
      Kt[jj * 72 + dd] = kvv;
      Vt[dd * 40 + jj] = vvv;
    }
    __syncthreads();

    // sim = Q(16x64) @ K^T : two 16-col subtiles, K-dim split 0-31 / 32-63
    v8f s0, s1;
#pragma unroll
    for (int e = 0; e < 8; ++e) { s0[e] = 0.f; s1[e] = 0.f; }
    s0 = wmma_f16(qa0, frag_b(Kt, 72), s0);
    s0 = wmma_f16(qa1, frag_b(Kt + 32, 72), s0);
    s1 = wmma_f16(qa0, frag_b(Kt + 16 * 72, 72), s1);
    s1 = wmma_f16(qa1, frag_b(Kt + 16 * 72 + 32, 72), s1);

    // bias + causal mask + online softmax (16-lane row reductions)
    float p0a[8], p1a[8];
#pragma unroll
    for (int e = 0; e < 8; ++e) {
      int row = i0 + e + 8 * hh;
      int c0 = j0 + cN, c1 = j0 + 16 + cN;
      float v0 = s0[e] * scale + bh[row * 1024 + c0];
      float v1 = s1[e] * scale + bh[row * 1024 + c1];
      if (c0 > row + 512) v0 = -3.0e38f;
      if (c1 > row + 512) v1 = -3.0e38f;
      float mx = fmaxf(v0, v1);
#pragma unroll
      for (int m = 1; m < 16; m <<= 1) mx = fmaxf(mx, __shfl_xor(mx, m, 32));
      float nm = fmaxf(rm[e], mx);
      float p0 = __expf(v0 - nm);
      float p1 = __expf(v1 - nm);
      float corr = __expf(rm[e] - nm);
      float ps = p0 + p1;
#pragma unroll
      for (int m = 1; m < 16; m <<= 1) ps += __shfl_xor(ps, m, 32);
      rl[e] = rl[e] * corr + ps;
      rm[e] = nm;
      p0a[e] = p0; p1a[e] = p1;
#pragma unroll
      for (int dt = 0; dt < 4; ++dt) o[dt][e] *= corr;
    }

    // stage P (16x32 f16) in this wave's private LDS slice, then P @ V
    _Float16* Pw = Pl + wv * 16 * 40;
#pragma unroll
    for (int e = 0; e < 8; ++e) {
      int rr = e + 8 * hh;
      Pw[rr * 40 + cN]      = (_Float16)p0a[e];
      Pw[rr * 40 + 16 + cN] = (_Float16)p1a[e];
    }
    // wave-private LDS RAW across lanes: drain the CDNA5 DS counter
    asm volatile("s_wait_dscnt 0" ::: "memory");
    v16h pa = frag_a(Pw, 40);
#pragma unroll
    for (int dt = 0; dt < 4; ++dt)
      o[dt] = wmma_f16(pa, frag_b(Vt + dt * 16 * 40, 40), o[dt]);
  }

  // normalize and write (b, w*512+row, h*64+d) f16 for the out-proj GEMM
#pragma unroll
  for (int e = 0; e < 8; ++e) {
    float inv = 1.0f / rl[e];
    int row = i0 + e + 8 * hh;
    long base = ((long)b * 4096 + w * 512 + row) * 1024 + h * 64;
#pragma unroll
    for (int dt = 0; dt < 4; ++dt)
      aout[base + dt * 16 + cN] = (_Float16)(o[dt][e] * inv);
  }
}

// ---------------------------------------------------------------------------
// launcher
// ---------------------------------------------------------------------------
extern "C" void kernel_launch(void* const* d_in, const int* in_sizes, int n_in,
                              void* d_out, int out_size, void* d_ws, size_t ws_size,
                              hipStream_t stream) {
  (void)in_sizes; (void)n_in; (void)out_size; (void)ws_size;
  const float* x     = (const float*)d_in[0];   // (2,4096,1024)
  const float* bias  = (const float*)d_in[1];   // (16,512,1024)
  const float* xlmem = (const float*)d_in[2];   // (2,2,16,512,64)
  const float* gamma = (const float*)d_in[3];   // (1024,)
  const float* wqkv  = (const float*)d_in[4];   // (1024,3072)
  const float* wout  = (const float*)d_in[5];   // (1024,1024)
  float* out = (float*)d_out;                   // (2,4096,1024) fp32
  float* mem = out + (size_t)8192 * 1024;       // memories (2,2,16,512,64) fp32

  char* ws = (char*)d_ws;
  _Float16* xn     = (_Float16*)(ws);                      // 16 MB
  _Float16* wqkv16 = (_Float16*)(ws + 16777216);           //  6 MB
  _Float16* wout16 = (_Float16*)(ws + 23068672);           //  2 MB
  _Float16* qb     = (_Float16*)(ws + 25165824);           // 16 MB
  _Float16* kb     = (_Float16*)(ws + 41943040);           // 16 MB
  _Float16* vb     = (_Float16*)(ws + 58720256);           // 16 MB
  _Float16* aout   = (_Float16*)(ws + 75497472);           // 16 MB

  hipLaunchKernelGGL(ln_kernel, dim3(8192), dim3(256), 0, stream, x, gamma, xn);
  hipLaunchKernelGGL(cvt_kernel, dim3(1024), dim3(256), 0, stream, wqkv, wqkv16, 1024 * 3072);
  hipLaunchKernelGGL(cvt_kernel, dim3(1024), dim3(256), 0, stream, wout, wout16, 1024 * 1024);
  hipLaunchKernelGGL(gemm_kernel, dim3(24, 64), dim3(256), 0, stream,
                     xn, wqkv16, 3072, 0, qb, kb, vb, mem, (float*)nullptr);
  hipLaunchKernelGGL(attn_kernel, dim3(512), dim3(512), 0, stream,
                     qb, kb, vb, xlmem, bias, aout);
  hipLaunchKernelGGL(gemm_kernel, dim3(8, 64), dim3(256), 0, stream,
                     aout, wout16, 1024, 1,
                     (_Float16*)nullptr, (_Float16*)nullptr, (_Float16*)nullptr,
                     (float*)nullptr, out);
}
